// CTCLoss_20504173871677
// MI455X (gfx1250) — compile-verified, matrix operand order checked by reading, not run on hardware
//
#include <hip/hip_runtime.h>
#include <stdint.h>

// CTC loss (warp-ctc semantics), MI455X / gfx1250.
// Phase 1: bandwidth-bound log-softmax + label gather (131 MB read once,
//          ~5.6 us at 23.3 TB/s), row sums via v_wmma_f32_16x16x32_f16.
// Phase 2: sequential alpha recursion, one wave32 per sample, async
//          global->LDS double buffering, cross-lane shuffles, no barriers.

#define CTC_T 1000
#define CTC_B 32
#define CTC_V 1024
#define CTC_L 100
#define CTC_S 201 /* 2L+1 */
#define NEGF (-1e30f)

#if defined(__has_builtin)
#if __has_builtin(__builtin_amdgcn_wmma_f32_16x16x32_f16)
#define HAVE_WMMA 1
#endif
#endif
#ifndef HAVE_WMMA
#define HAVE_WMMA 0
#endif

typedef _Float16 v16h __attribute__((ext_vector_type(16)));
typedef float v8f __attribute__((ext_vector_type(8)));

// ---------- CDNA5 async global->LDS (gfx1250 VGLOBAL async ops) ----------
__device__ __forceinline__ uint32_t lds_off32(const void* p) {
  // low 32 bits of a generic shared pointer == LDS byte offset (ISA aperture rule)
  return (uint32_t)(uintptr_t)p;
}
__device__ __forceinline__ void async_load_b128(uint32_t lds_byte_off, uint64_t gaddr) {
  asm volatile("global_load_async_to_lds_b128 %0, %1, off"
               :: "v"(lds_byte_off), "v"(gaddr) : "memory");
}
__device__ __forceinline__ void wait_asynccnt0() {
  asm volatile("s_wait_asynccnt 0" ::: "memory");
}

// ---------- log-sum-exp helpers (finite -inf surrogate like reference) ----------
__device__ __forceinline__ float logaddexp2_(float a, float b) {
  float m = fmaxf(a, b);
  float d = fminf(a, b) - m;
  return m + log1pf(__expf(d));
}
__device__ __forceinline__ float logaddexp3_(float a, float b, float c) {
  float m = fmaxf(a, fmaxf(b, c));
  return m + __logf(__expf(a - m) + __expf(b - m) + __expf(c - m));
}

// =====================================================================
// Kernel 1: log-softmax LSE + gather of extended-label log-probs.
// 16 rows (one (t,b) row = 1024 f32 = 4KB) per 256-thread block.
// =====================================================================
#define ROWS_PER_BLK 16
#define ROW_PITCH 1028 /* floats: 4112B pitch, 16B aligned, de-conflicts column reads */

__global__ __launch_bounds__(256) void ctc_lse_gather(
    const float* __restrict__ acts, const int* __restrict__ labels,
    float* __restrict__ lp_out) {
  __shared__ __attribute__((aligned(16))) float rows[ROWS_PER_BLK * ROW_PITCH];
  __shared__ float s_rowmax[ROWS_PER_BLK];
  __shared__ float s_lse[ROWS_PER_BLK];
  __shared__ float s_part[8 * 16];

  const int tid = threadIdx.x;
  const int lane = tid & 31;
  const int wave = tid >> 5;
  const int rowBase = blockIdx.x * ROWS_PER_BLK;

  // ---- async-stage 16 contiguous rows (64KB) into LDS with per-row padding
  {
    const uint64_t gbase = (uint64_t)(uintptr_t)(acts + (size_t)rowBase * CTC_V);
    const uint32_t lbase = lds_off32(rows);
#pragma unroll
    for (int j = 0; j < 16; ++j) {
      const int chunk = tid + j * 256;  // 16B chunks, 0..4095 (256 per row)
      const int r = chunk >> 8;
      const int w = chunk & 255;
      async_load_b128(lbase + (uint32_t)(r * (ROW_PITCH * 4) + w * 16),
                      gbase + (uint64_t)chunk * 16u);
    }
  }
  wait_asynccnt0();
  __syncthreads();

  // ---- per-row max: 16 threads per row, shuffle tree within 16-lane group
  {
    const int r = tid >> 4;
    const int sub = tid & 15;
    float m = -3.4e38f;
#pragma unroll 8
    for (int j = 0; j < 64; ++j)
      m = fmaxf(m, rows[r * ROW_PITCH + sub + j * 16]);
#pragma unroll
    for (int off = 8; off; off >>= 1)
      m = fmaxf(m, __shfl_xor(m, off, 32));
    if (sub == 0) s_rowmax[r] = m;
  }
  __syncthreads();

  // ---- row sums of exp(x - max)
  const int mrow = lane & 15;  // M index of A matrix
  const int half = lane >> 4;  // A-layout half (K groups)
  const float rm = s_rowmax[mrow];
#if HAVE_WMMA
  {
    // Each wave: A = 16 rows x 32 cols (f16 exp values), B = ones -> D col0 = row sums.
    v8f acc = {0.f, 0.f, 0.f, 0.f, 0.f, 0.f, 0.f, 0.f};
    v16h ones;
#pragma unroll
    for (int i = 0; i < 16; ++i) ones[i] = (_Float16)1.0f;
    const float* rp = &rows[mrow * ROW_PITCH];
#pragma unroll
    for (int c = 0; c < 4; ++c) {
      const int base = wave * 128 + c * 32;
      v16h a;
#pragma unroll
      for (int i = 0; i < 8; ++i)  // K = base + half*8 .. +7
        a[i] = (_Float16)__expf(rp[base + half * 8 + i] - rm);
#pragma unroll
      for (int i = 0; i < 8; ++i)  // K = base + 16 + half*8 .. +7
        a[8 + i] = (_Float16)__expf(rp[base + 16 + half * 8 + i] - rm);
      acc = __builtin_amdgcn_wmma_f32_16x16x32_f16(false, a, false, ones,
                                                   (short)0, acc, false, false);
    }
    // D col N=0 lives in lane0 (M=0..7 over 8 VGPRs) and lane16 (M=8..15)
    if (lane == 0) {
#pragma unroll
      for (int i = 0; i < 8; ++i) s_part[wave * 16 + i] = acc[i];
    } else if (lane == 16) {
#pragma unroll
      for (int i = 0; i < 8; ++i) s_part[wave * 16 + 8 + i] = acc[i];
    }
  }
#else
  {
    const int r = tid >> 4;
    const int sub = tid & 15;
    const float rm2 = s_rowmax[r];
    float p = 0.f;
#pragma unroll 8
    for (int j = 0; j < 64; ++j)
      p += __expf(rows[r * ROW_PITCH + sub + j * 16] - rm2);
#pragma unroll
    for (int off = 8; off; off >>= 1)
      p += __shfl_xor(p, off, 32);
    if (sub == 0) s_part[r] = p;
  }
#endif
  __syncthreads();

  if (tid < 16) {
#if HAVE_WMMA
    float s = 0.f;
#pragma unroll
    for (int w = 0; w < 8; ++w) s += s_part[w * 16 + tid];
#else
    float s = s_part[tid];
#endif
    s_lse[tid] = s_rowmax[tid] + __logf(s);
  }
  __syncthreads();

  // ---- gather extended-label log-probs: lp_ext[row, s] = row[ext(s)] - lse
#pragma unroll 1
  for (int r = 0; r < ROWS_PER_BLK; ++r) {
    if (tid < CTC_S) {
      const int b = (rowBase + r) & (CTC_B - 1);
      int colv = 0;
      if (tid & 1) colv = labels[b * CTC_L + (tid >> 1)];
      lp_out[(size_t)(rowBase + r) * CTC_S + tid] =
          rows[r * ROW_PITCH + colv] - s_lse[r];
    }
  }
}

// =====================================================================
// Kernel 2: forward alpha recursion. One wave32 per sample, 7 states/lane
// (7*32=224 >= S=201), neighbors via shuffles, lp tiles double-buffered
// in LDS via async loads (no barriers on the 1000-step critical path).
// =====================================================================
#define KT 20   /* t-rows per LDS tile */
#define LPP 208 /* floats per lp row in LDS: 832B = 52 x 16B chunks */
#define TILE_CHUNKS (KT * 52)

__global__ __launch_bounds__(32) void ctc_alpha_scan(
    const float* __restrict__ lp_ext, const int* __restrict__ labels,
    const int* __restrict__ act_lens, const int* __restrict__ label_lens,
    float* __restrict__ costs) {
  __shared__ __attribute__((aligned(16))) float tile[2][KT * LPP];
  __shared__ float s_alpha[224];

  const int b = blockIdx.x;
  const int lane = threadIdx.x;

  float a[7];
  int colv[7];
  bool act[7], skipok[7];
#pragma unroll
  for (int j = 0; j < 7; ++j) {
    const int s = lane * 7 + j;
    act[j] = (s < CTC_S);
    colv[j] = 0;
    skipok[j] = false;
    if (act[j] && (s & 1)) {
      const int k = s >> 1;
      const int l0 = labels[b * CTC_L + k];
      colv[j] = l0;
      if (s >= 3) skipok[j] = (l0 != labels[b * CTC_L + k - 1]);
    }
    a[j] = NEGF;
  }
  (void)colv;
  const int alen = act_lens[b];
  const uint32_t lt[2] = {lds_off32(&tile[0][0]), lds_off32(&tile[1][0])};

  auto issue_tile = [&](int buf, int t0) {
    const uint32_t lb = lt[buf];
#pragma unroll 1
    for (int i = 0; i < (TILE_CHUNKS + 31) / 32; ++i) {
      const int c = lane + i * 32;
      if (c < TILE_CHUNKS) {
        const int r = c / 52;
        const int w = c - r * 52;
        const uint64_t g = (uint64_t)(uintptr_t)lp_ext +
                           (((size_t)(t0 + r) * CTC_B + b) * CTC_S) * 4ull +
                           (uint64_t)w * 16u;
        async_load_b128(lb + (uint32_t)(r * (LPP * 4) + w * 16), g);
      }
    }
  };

  issue_tile(0, 0);
  wait_asynccnt0();
  int buf = 0;
#pragma unroll 1
  for (int t0 = 0; t0 < CTC_T; t0 += KT) {
    if (t0 + KT < CTC_T) issue_tile(buf ^ 1, t0 + KT);  // prefetch next tile
    const float* tp = &tile[buf][0];
#pragma unroll 1
    for (int tt = 0; tt < KT; ++tt) {
      const int t = t0 + tt;
      float lp[7];
#pragma unroll
      for (int j = 0; j < 7; ++j)
        lp[j] = act[j] ? tp[tt * LPP + lane * 7 + j] : NEGF;
      if (t == 0) {
#pragma unroll
        for (int j = 0; j < 7; ++j)
          a[j] = (act[j] && (lane * 7 + j) < 2) ? lp[j] : NEGF;
      } else {
        float p6 = __shfl_up(a[6], 1, 32);  // alpha[s-1] across lane boundary
        float p5 = __shfl_up(a[5], 1, 32);  // alpha[s-2] across lane boundary
        if (lane == 0) { p6 = NEGF; p5 = NEGF; }
        float n[7];
        n[0] = lp[0] + logaddexp3_(a[0], p6, skipok[0] ? p5 : NEGF);
        n[1] = lp[1] + logaddexp3_(a[1], a[0], skipok[1] ? p6 : NEGF);
#pragma unroll
        for (int j = 2; j < 7; ++j)
          n[j] = lp[j] + logaddexp3_(a[j], a[j - 1], skipok[j] ? a[j - 2] : NEGF);
        const bool frz = (t >= alen);  // variable-length freeze (reference semantics)
#pragma unroll
        for (int j = 0; j < 7; ++j)
          a[j] = act[j] ? (frz ? a[j] : n[j]) : NEGF;
      }
    }
    wait_asynccnt0();
    buf ^= 1;
  }

  __syncthreads();  // single-wave WG: effectively s_nop, orders LDS ops
#pragma unroll
  for (int j = 0; j < 7; ++j) s_alpha[lane * 7 + j] = a[j];
  __syncthreads();
  if (lane == 0) {
    const int sl = 2 * label_lens[b];
    costs[b] = -logaddexp2_(s_alpha[sl], s_alpha[sl - 1]);
  }
}

// =====================================================================
// Kernel 3: deterministic fixed-order sum of per-sample costs.
// =====================================================================
__global__ void ctc_sum_costs(const float* __restrict__ costs,
                              float* __restrict__ out) {
  if (threadIdx.x == 0 && blockIdx.x == 0) {
    float s = 0.f;
#pragma unroll
    for (int b = 0; b < CTC_B; ++b) s += costs[b];
    out[0] = s;
  }
}

// =====================================================================
// Fallback: fully fused (softmax + recursion) per-sample kernel, used only
// if d_ws cannot hold the 25.7MB lp_ext table. 256 threads / sample.
// =====================================================================
__global__ __launch_bounds__(256) void ctc_fused(
    const float* __restrict__ acts, const int* __restrict__ labels,
    const int* __restrict__ act_lens, const int* __restrict__ label_lens,
    float* __restrict__ costs) {
  __shared__ float row[CTC_V];
  __shared__ float abuf[2][256];
  __shared__ float red[8];
  __shared__ float s_lse;
  const int b = blockIdx.x, tid = threadIdx.x;
  const int lane = tid & 31, wid = tid >> 5;
  const bool active = tid < CTC_S;
  int colv = 0;
  bool skipok = false;
  if (active && (tid & 1)) {
    const int k = tid >> 1;
    const int l0 = labels[b * CTC_L + k];
    colv = l0;
    if (tid >= 3) skipok = (l0 != labels[b * CTC_L + k - 1]);
  }
  const int alen = act_lens[b];
  int cur = 0;
  for (int t = 0; t < CTC_T; ++t) {
    const float* ar = acts + ((size_t)t * CTC_B + b) * CTC_V;
    float v0 = ar[tid], v1 = ar[tid + 256], v2 = ar[tid + 512], v3 = ar[tid + 768];
    row[tid] = v0; row[tid + 256] = v1; row[tid + 512] = v2; row[tid + 768] = v3;
    float m = fmaxf(fmaxf(v0, v1), fmaxf(v2, v3));
#pragma unroll
    for (int o = 16; o; o >>= 1) m = fmaxf(m, __shfl_xor(m, o, 32));
    if (lane == 0) red[wid] = m;
    __syncthreads();
    const float rmax = fmaxf(fmaxf(fmaxf(red[0], red[1]), fmaxf(red[2], red[3])),
                             fmaxf(fmaxf(red[4], red[5]), fmaxf(red[6], red[7])));
    __syncthreads();  // everyone consumed red before reuse
    float p = __expf(v0 - rmax) + __expf(v1 - rmax) + __expf(v2 - rmax) + __expf(v3 - rmax);
#pragma unroll
    for (int o = 16; o; o >>= 1) p += __shfl_xor(p, o, 32);
    if (lane == 0) red[wid] = p;
    __syncthreads();
    if (tid == 0) {
      const float ss = red[0] + red[1] + red[2] + red[3] + red[4] + red[5] + red[6] + red[7];
      s_lse = rmax + __logf(ss);
    }
    __syncthreads();
    const float lpv = active ? (row[colv] - s_lse) : NEGF;
    float nv;
    if (t == 0) {
      nv = (active && tid < 2) ? lpv : NEGF;
    } else {
      const float am = abuf[cur][tid];
      const float a1 = (tid >= 1) ? abuf[cur][tid - 1] : NEGF;
      const float a2 = skipok ? abuf[cur][tid - 2] : NEGF;
      nv = lpv + logaddexp3_(am, a1, a2);
      if (t >= alen) nv = am;
      if (!active) nv = NEGF;
    }
    const int nb = (t == 0) ? 0 : (cur ^ 1);
    abuf[nb][tid] = nv;
    cur = nb;
    __syncthreads();
  }
  if (tid == 0) {
    const int sl = 2 * label_lens[b];
    costs[b] = -logaddexp2_(abuf[cur][sl], abuf[cur][sl - 1]);
  }
}

extern "C" void kernel_launch(void* const* d_in, const int* in_sizes, int n_in,
                              void* d_out, int out_size, void* d_ws, size_t ws_size,
                              hipStream_t stream) {
  (void)in_sizes; (void)n_in; (void)out_size;
  const float* acts = (const float*)d_in[0];
  const int* labels = (const int*)d_in[1];
  const int* act_lens = (const int*)d_in[2];
  const int* label_lens = (const int*)d_in[3];
  float* out = (float*)d_out;

  const size_t lp_elems = (size_t)CTC_T * CTC_B * CTC_S;
  const size_t need = (lp_elems + CTC_B) * sizeof(float);
  if (ws_size >= need) {
    float* lp = (float*)d_ws;
    float* costs = lp + lp_elems;
    ctc_lse_gather<<<dim3((CTC_T * CTC_B) / ROWS_PER_BLK), dim3(256), 0, stream>>>(
        acts, labels, lp);
    ctc_alpha_scan<<<dim3(CTC_B), dim3(32), 0, stream>>>(
        lp, labels, act_lens, label_lens, costs);
    ctc_sum_costs<<<dim3(1), dim3(1), 0, stream>>>(costs, out);
  } else if (ws_size >= CTC_B * sizeof(float)) {
    float* costs = (float*)d_ws;
    ctc_fused<<<dim3(CTC_B), dim3(256), 0, stream>>>(
        acts, labels, act_lens, label_lens, costs);
    ctc_sum_costs<<<dim3(1), dim3(1), 0, stream>>>(costs, out);
  }
}
